// Attention_54872502174163
// MI455X (gfx1250) — matrix operand order repaired and checked by reading, not compiled, not run on hardware
//
#include <hip/hip_runtime.h>
#include <hip/hip_bf16.h>

// ---------------------------------------------------------------------------
// Types for CDNA5 WMMA (wave32, bf16 16x16x32, f32 accumulate)
// ---------------------------------------------------------------------------
typedef __bf16 bf16_t;
typedef __attribute__((ext_vector_type(16))) __bf16        v16bf;
typedef __attribute__((ext_vector_type(8)))  float         v8f;
typedef __attribute__((ext_vector_type(4)))  unsigned int  u32x4;

union FragBF {
    v16bf bf;
    u32x4 q[2];
};

__device__ __forceinline__ bf16_t f2bf(float f) {
    unsigned u = __builtin_bit_cast(unsigned, f);
    u += 0x7FFFu + ((u >> 16) & 1u);          // round-to-nearest-even
    unsigned short h = (unsigned short)(u >> 16);
    return __builtin_bit_cast(bf16_t, h);
}

__device__ __forceinline__ v8f wmma_bf16(v16bf a, v16bf b, v8f c) {
    return __builtin_amdgcn_wmma_f32_16x16x32_bf16(
        false, a, false, b, (short)0, c, false, false);
}

// ---------------------------------------------------------------------------
// Async global->LDS copy (CDNA5 GLOBAL_LOAD_ASYNC_TO_LDS_B128, ASYNCcnt).
// Probe-learned signature: param0 = v4i* in AS(1) (printed '__device__'),
// param1 = LDS side, then imm offset + imm cpol.
// ---------------------------------------------------------------------------
#if defined(__has_builtin)
#if __has_builtin(__builtin_amdgcn_global_load_async_to_lds_b128)
#define HAVE_ASYNC_LDS 1
#endif
#endif
#ifndef HAVE_ASYNC_LDS
#define HAVE_ASYNC_LDS 0
#endif

typedef int v4i_vs __attribute__((vector_size(16)));
typedef __attribute__((address_space(1))) v4i_vs* gptr_v4i;
typedef __attribute__((address_space(3))) v4i_vs* lptr_v4i;

__device__ __forceinline__ void async_copy16(const bf16_t* g, bf16_t* l) {
#if HAVE_ASYNC_LDS
    __builtin_amdgcn_global_load_async_to_lds_b128((gptr_v4i)g, (lptr_v4i)l, 0, 0);
#else
    *(u32x4*)l = *(const u32x4*)g;            // sync fallback (vmem + ds)
#endif
}

template <int N>
__device__ __forceinline__ void wait_async() {
#if HAVE_ASYNC_LDS
#if __has_builtin(__builtin_amdgcn_s_wait_asynccnt)
    __builtin_amdgcn_s_wait_asynccnt(N);
#else
    asm volatile("s_wait_asynccnt %0" :: "i"(N) : "memory");
#endif
#endif
}

// Problem constants
#define BB   2
#define TT   2048
#define CC   1024
#define HH   16
#define HD   64
#define RR   8
#define LORA_SCALE 2.0f

// ---------------------------------------------------------------------------
// Fold LoRA into the dense weight: Weff[o,c] = W[o,c] + 2 * sum_r B[o,r]A[r,c]
// ---------------------------------------------------------------------------
__global__ void __launch_bounds__(256)
fold_weight_kernel(const float* __restrict__ W, const float* __restrict__ A,
                   const float* __restrict__ Bl, bf16_t* __restrict__ out,
                   int O, int K) {
    int idx = blockIdx.x * 256 + threadIdx.x;
    if (idx >= O * K) return;
    int o = idx / K, c = idx - o * K;
    float acc = W[idx];
    #pragma unroll
    for (int r = 0; r < RR; ++r)
        acc += LORA_SCALE * Bl[o * RR + r] * A[r * K + c];
    out[idx] = f2bf(acc);
}

__global__ void __launch_bounds__(256)
cvt_bf16_kernel(const float* __restrict__ in, bf16_t* __restrict__ out, int n) {
    int idx = blockIdx.x * 256 + threadIdx.x;
    if (idx < n) out[idx] = f2bf(in[idx]);
}

// ---------------------------------------------------------------------------
// GEMM: out[M,N] = Xb[M,K] @ Wb[N,K]^T + bias[N]
// Block: 256 threads = 8 waves -> 128 rows x 64 cols.
// W tile (64x32 bf16 = 4KB = 256 x B128) is async-staged into LDS per k-step,
// double-buffered so the next copy overlaps current WMMAs; all 8 waves share
// the staged B-fragments (dedups 8x redundant global traffic).
// ---------------------------------------------------------------------------
template <bool STORE_F32>
__global__ void __launch_bounds__(256)
gemm_bias_kernel(const bf16_t* __restrict__ X, const bf16_t* __restrict__ W,
                 const float* __restrict__ bias, bf16_t* __restrict__ outb,
                 float* __restrict__ outf, int M, int N, int K) {
    __shared__ bf16_t Wt[2][64][32];          // [buf][n within tile][k chunk]

    const int lane = threadIdx.x & 31;
    const int wid  = threadIdx.x >> 5;
    const int m    = lane & 15;
    const int hi   = lane >> 4;
    const int rowBase = blockIdx.x * 128 + wid * 16;
    const int colBase = blockIdx.y * 64;

    // staging coordinates: one B128 per thread per step
    const int sr = threadIdx.x >> 2;          // 0..63 col within tile
    const int sc = (threadIdx.x & 3) * 8;     // 0,8,16,24 k offset
    const bf16_t* wsrc = W + (size_t)(colBase + sr) * K + sc;

    const bf16_t* xrow = X + (size_t)(rowBase + m) * K;

    v8f acc[4] = {};
    async_copy16(wsrc, &Wt[0][sr][sc]);       // stage k-step 0
    int buf = 0;
    for (int kk = 0; kk < K; kk += 32) {
        if (kk + 32 < K) {                    // prefetch next step, then wait
            async_copy16(wsrc + kk + 32, &Wt[buf ^ 1][sr][sc]);
            wait_async<1>();                  // oldest (current buf) done
        } else {
            wait_async<0>();
        }
        __syncthreads();

        FragBF a;
        const int c0 = kk + hi * 16;          // A chunks {0..15}/{16..31}
        a.q[0] = *(const u32x4*)(xrow + c0);
        a.q[1] = *(const u32x4*)(xrow + c0 + 8);
        #pragma unroll
        for (int t = 0; t < 4; ++t) {
            FragBF b;
            const bf16_t* wrow = &Wt[buf][t * 16 + m][hi * 16];
            b.q[0] = *(const u32x4*)(wrow);
            b.q[1] = *(const u32x4*)(wrow + 8);
            acc[t] = wmma_bf16(a.bf, b.bf, acc[t]);
        }
        __syncthreads();                      // tile reads done before reuse
        buf ^= 1;
    }

    #pragma unroll
    for (int t = 0; t < 4; ++t) {
        const int col = colBase + t * 16 + m;
        const float bv = bias[col];
        #pragma unroll
        for (int j = 0; j < 8; ++j) {
            const int row = rowBase + j + hi * 8;
            const float v = acc[t][j] + bv;
            if (STORE_F32) outf[(size_t)row * N + col] = v;
            else           outb[(size_t)row * N + col] = f2bf(v);
        }
    }
}

// ---------------------------------------------------------------------------
// Flash attention (causal), bf16 QKV from fused qkv buffer [B*T, 3C].
// Grid: (T/128, B*H). Block: 256 threads = 8 waves; wave -> 16 q rows.
// Per 32-key step: K tile async-staged row-major into LDS (shared by all 8
// waves, no transpose needed for Kt B-fragments); V staged transposed via
// registers; P transposed per-wave through LDS (C-layout -> A-layout).
// ---------------------------------------------------------------------------
__global__ void __launch_bounds__(256)
flash_attn_kernel(const bf16_t* __restrict__ qkv, bf16_t* __restrict__ att) {
    __shared__ bf16_t Kt[32][HD];        // K tile row-major: Kt[k][d]
    __shared__ bf16_t Vt[HD][32];        // transposed V tile: Vt[d][k]
    __shared__ bf16_t Pb[8][16][32];     // per-wave P tile: Pb[w][m][k]

    const int lane = threadIdx.x & 31;
    const int wid  = threadIdx.x >> 5;
    const int m    = lane & 15;
    const int hi   = lane >> 4;
    const int qb   = blockIdx.x;
    const int bh   = blockIdx.y;
    const int b    = bh >> 4;
    const int h    = bh & 15;
    const int qBase = qb * 128 + wid * 16;
    const size_t rowStride = 3 * CC;

    // Q A-fragments (d-chunks 0..31 and 32..63), loaded once per wave
    const bf16_t* qptr = qkv + (size_t)(b * TT + qBase + m) * rowStride + h * HD;
    FragBF qf[2];
    #pragma unroll
    for (int f = 0; f < 2; ++f) {
        const int c0 = f * 32 + hi * 16;
        qf[f].q[0] = *(const u32x4*)(qptr + c0);
        qf[f].q[1] = *(const u32x4*)(qptr + c0 + 8);
    }

    v8f o[4] = {};
    float mrow[8], lrow[8];
    #pragma unroll
    for (int j = 0; j < 8; ++j) { mrow[j] = -1e30f; lrow[j] = 0.0f; }

    // per-thread staging coords
    const int kr = threadIdx.x >> 3;          // 0..31 key row
    const int kc = (threadIdx.x & 7) * 8;     // 0..56 d offset (K stage)
    const int vk = threadIdx.x >> 3;          // 0..31 key row (V stage)
    const int vd = (threadIdx.x & 7) * 8;     // d group (V stage)

    const int nsteps = (qb + 1) * 4;          // uniform across block (causal)
    for (int s = 0; s < nsteps; ++s) {
        const int kb = s * 32;

        // ---- async stage of K tile (row-major, byte-for-byte) ----
        async_copy16(qkv + (size_t)(b * TT + kb + kr) * rowStride + CC + h * HD + kc,
                     &Kt[kr][kc]);
        // ---- cooperative stage of V tile, transposed via registers ----
        {
            const bf16_t* vsrc = qkv + (size_t)(b * TT + kb + vk) * rowStride
                                 + 2 * CC + h * HD + vd;
            u32x4 vv = *(const u32x4*)vsrc;
            const bf16_t* ve = (const bf16_t*)&vv;
            #pragma unroll
            for (int e = 0; e < 8; ++e) Vt[vd + e][vk] = ve[e];
        }
        wait_async<0>();
        __syncthreads();

        if (kb <= qBase + 15) {               // wave-uniform causal bound
            // ---- S = (Q K^T) * 1/sqrt(hd), two 16x16 key tiles ----
            v8f sfr[2];
            #pragma unroll
            for (int t = 0; t < 2; ++t) {
                sfr[t] = {};
                #pragma unroll
                for (int f = 0; f < 2; ++f) {
                    FragBF kf;
                    const bf16_t* kp = &Kt[t * 16 + m][f * 32 + hi * 16];
                    kf.q[0] = *(const u32x4*)(kp);
                    kf.q[1] = *(const u32x4*)(kp + 8);
                    sfr[t] = wmma_bf16(qf[f].bf, kf.bf, sfr[t]);
                }
            }
            // scale + causal mask
            #pragma unroll
            for (int t = 0; t < 2; ++t) {
                const int kg = kb + t * 16 + m;
                #pragma unroll
                for (int j = 0; j < 8; ++j) {
                    const int qg = qBase + j + hi * 8;
                    const float v = sfr[t][j] * 0.125f;
                    sfr[t][j] = (kg > qg) ? -1e30f : v;
                }
            }
            // ---- online softmax per q-row (VGPR index j) ----
            #pragma unroll
            for (int j = 0; j < 8; ++j) {
                float cmax = fmaxf(sfr[0][j], sfr[1][j]);
                cmax = fmaxf(cmax, __shfl_xor(cmax, 1, 32));
                cmax = fmaxf(cmax, __shfl_xor(cmax, 2, 32));
                cmax = fmaxf(cmax, __shfl_xor(cmax, 4, 32));
                cmax = fmaxf(cmax, __shfl_xor(cmax, 8, 32));
                const float mn = fmaxf(mrow[j], cmax);
                const float sc = __expf(mrow[j] - mn);
                const float p0 = __expf(sfr[0][j] - mn);
                const float p1 = __expf(sfr[1][j] - mn);
                sfr[0][j] = p0; sfr[1][j] = p1;
                float rs = p0 + p1;
                rs += __shfl_xor(rs, 1, 32);
                rs += __shfl_xor(rs, 2, 32);
                rs += __shfl_xor(rs, 4, 32);
                rs += __shfl_xor(rs, 8, 32);
                lrow[j] = lrow[j] * sc + rs;
                mrow[j] = mn;
                #pragma unroll
                for (int t = 0; t < 4; ++t) o[t][j] *= sc;
            }
            // ---- P: C-layout regs -> LDS -> A-layout fragment ----
            #pragma unroll
            for (int t = 0; t < 2; ++t)
                #pragma unroll
                for (int j = 0; j < 8; ++j)
                    Pb[wid][j + hi * 8][t * 16 + m] = f2bf(sfr[t][j]);
            FragBF pf;
            {
                const int c0 = hi * 16;
                pf.q[0] = *(const u32x4*)&Pb[wid][m][c0];
                pf.q[1] = *(const u32x4*)&Pb[wid][m][c0 + 8];
            }
            // ---- O += P @ V (four 16-wide d chunks) ----
            #pragma unroll
            for (int t = 0; t < 4; ++t) {
                FragBF vf;
                const int d = t * 16 + m;
                vf.q[0] = *(const u32x4*)&Vt[d][hi * 16];
                vf.q[1] = *(const u32x4*)&Vt[d][hi * 16 + 8];
                o[t] = wmma_bf16(pf.bf, vf.bf, o[t]);
            }
        }
        __syncthreads();
    }

    // epilogue: normalize and store bf16 to att[B*T, C]
    float linv[8];
    #pragma unroll
    for (int j = 0; j < 8; ++j) linv[j] = 1.0f / lrow[j];
    #pragma unroll
    for (int t = 0; t < 4; ++t)
        #pragma unroll
        for (int j = 0; j < 8; ++j) {
            const int row = b * TT + qBase + j + hi * 8;
            const int col = h * HD + t * 16 + m;
            att[(size_t)row * CC + col] = f2bf(o[t][j] * linv[j]);
        }
}

// ---------------------------------------------------------------------------
// Launch
// ---------------------------------------------------------------------------
extern "C" void kernel_launch(void* const* d_in, const int* in_sizes, int n_in,
                              void* d_out, int out_size, void* d_ws, size_t ws_size,
                              hipStream_t stream) {
    const float* x      = (const float*)d_in[0];
    const float* W_attn = (const float*)d_in[1];
    const float* b_attn = (const float*)d_in[2];
    const float* A_attn = (const float*)d_in[3];
    const float* B_attn = (const float*)d_in[4];
    const float* W_proj = (const float*)d_in[5];
    const float* b_proj = (const float*)d_in[6];
    const float* A_proj = (const float*)d_in[7];
    const float* B_proj = (const float*)d_in[8];

    const int M = BB * TT;                 // 4096 token rows
    char* ws = (char*)d_ws;
    bf16_t* xb  = (bf16_t*)(ws);                              //  8 MB
    bf16_t* wa  = (bf16_t*)(ws + (size_t)8  * 1024 * 1024);   //  6 MB
    bf16_t* wp  = (bf16_t*)(ws + (size_t)14 * 1024 * 1024);   //  2 MB
    bf16_t* qkv = (bf16_t*)(ws + (size_t)16 * 1024 * 1024);   // 24 MB
    bf16_t* att = (bf16_t*)(ws + (size_t)40 * 1024 * 1024);   //  8 MB

    fold_weight_kernel<<<(3 * CC * CC + 255) / 256, 256, 0, stream>>>(
        W_attn, A_attn, B_attn, wa, 3 * CC, CC);
    fold_weight_kernel<<<(CC * CC + 255) / 256, 256, 0, stream>>>(
        W_proj, A_proj, B_proj, wp, CC, CC);
    cvt_bf16_kernel<<<(M * CC + 255) / 256, 256, 0, stream>>>(x, xb, M * CC);

    gemm_bias_kernel<false><<<dim3(M / 128, (3 * CC) / 64), 256, 0, stream>>>(
        xb, wa, b_attn, qkv, nullptr, M, 3 * CC, CC);

    flash_attn_kernel<<<dim3(TT / 128, BB * HH), 256, 0, stream>>>(qkv, att);

    gemm_bias_kernel<true><<<dim3(M / 128, CC / 64), 256, 0, stream>>>(
        att, wp, b_proj, nullptr, (float*)d_out, M, CC, CC);
}
